// HybridLlamaAttention_55027120996624
// MI455X (gfx1250) — compile-verified, matrix operand order checked by reading, not run on hardware
//
#include <hip/hip_runtime.h>

// ---------------------------------------------------------------------------
// Gemma2-style attention block on gfx1250 (MI455X), bf16 WMMA pipeline.
//   x:[1,2048,2048] f32 ; wq/wk/wv/wo:[2048,2048] f32 ; out:[1,2048,2048] f32
// Pipeline (all bf16 matrix math, f32 accum):
//   cvt:   f32 -> bf16 (x, wq, wk, wv, wo)
//   gemm<1>: Q = x*wq^T  (+fused RoPE)  -> qh [h][s][128] bf16
//   gemm<1>: K = x*wk^T  (+fused RoPE)  -> kh [h][s][128] bf16
//   gemm<2>: V = x*wv^T  (transposed)   -> vt [h][128][s] bf16
//   attn:  flash attention w/ tanh softcap + causal mask -> ob [s][2048] bf16
//   gemm<0>: out = ob*wo^T -> d_out f32
// GEMM B-tiles are staged into LDS by the Tensor Data Mover
// (tensor_load_to_lds + s_wait_tensorcnt), double-buffered.
// ---------------------------------------------------------------------------

#define SEQ 2048
#define DIM 2048
#define NH  16
#define HD  128

typedef __attribute__((ext_vector_type(16))) __bf16        v16bf;
typedef __attribute__((ext_vector_type(8)))  float         v8f;
typedef __attribute__((ext_vector_type(4)))  unsigned int  v4u;
typedef __attribute__((ext_vector_type(8)))  int           v8i;
typedef __attribute__((ext_vector_type(4)))  int           v4i;

union Frag {            // one WMMA 16x16x32 bf16 operand per lane: 32 bytes
    v16bf v;
    uint4 u[2];
};

__device__ __forceinline__ unsigned short f2bf(float f) {
    union { float f; unsigned int u; } x;
    x.f = f;
    unsigned int r = x.u + 0x7FFFu + ((x.u >> 16) & 1u);  // round-to-nearest-even
    return (unsigned short)(r >> 16);
}

// TDM: async-load a [128 rows x 32 k] bf16 tile of a row-major [2048 x 2048]
// bf16 tensor into LDS (rows contiguous: LDS layout [128][32]).
// D# per CDNA5 ISA ch.8: group0 {count=1, lds_addr, global_addr, type=2};
// group1 {data_size=2B, tensor_dim0=2048, tensor_dim1=2048, tile_dim0=32,
// tile_dim1=128, tensor_dim0_stride=2048}; groups 2/3 zero (<=2D tensor).
__device__ __forceinline__ void tdm_load_tile_128x32(
    const unsigned short* gsrc, unsigned lds_off) {
    unsigned long long ga = (unsigned long long)(size_t)gsrc;
    v4u g0;
    g0[0] = 1u;                                   // count=1, user descriptor
    g0[1] = lds_off;                              // lds_addr (bytes)
    g0[2] = (unsigned)ga;                         // global_addr[31:0]
    g0[3] = (unsigned)((ga >> 32) & 0x1FFFFFFu)   // global_addr[56:32]
          | (2u << 30);                           // type=2 ("image")
    v8i g1;
    g1[0] = (int)(1u << 16);       // workgroup_mask=0, data_size=1 (2 bytes)
    g1[1] = (int)0x08000000u;      // tensor_dim0[15:0]=2048 in bits 63:48
    g1[2] = (int)0x08000000u;      // tensor_dim1[15:0]=2048 in bits 95:80
    g1[3] = (int)(32u << 16);      // tile_dim0=32 in bits 127:112
    g1[4] = 128;                   // tile_dim1=128 (tile_dim2=0)
    g1[5] = 2048;                  // tensor_dim0_stride=2048 elements
    g1[6] = 0;                     // stride hi / tensor_dim1_stride
    g1[7] = 0;
    v4i z4 = {0, 0, 0, 0};
#if defined(__clang_major__) && (__clang_major__ >= 23)
    v8i z8 = {0, 0, 0, 0, 0, 0, 0, 0};
    __builtin_amdgcn_tensor_load_to_lds(g0, g1, z4, z4, z8, 0);
#else
    __builtin_amdgcn_tensor_load_to_lds(g0, g1, z4, z4, 0);
#endif
}

// ---------------------------------------------------------------------------
__global__ void cvt_f32_bf16(const float* __restrict__ in,
                             unsigned short* __restrict__ out, int n) {
    int i = blockIdx.x * blockDim.x + threadIdx.x;
    if (i < n) out[i] = f2bf(in[i]);
}

// ---------------------------------------------------------------------------
// Tiled GEMM: D[M,N] = A[M,K] * B[N,K]^T   (M=SEQ, N=K=DIM), bf16 in, f32 acc.
// 128 threads = 4 waves. Tile: BM=64 (16 rows/wave), BN=128, BK=32.
// B tiles staged to LDS by the TDM (wave 0 issues), double-buffered.
// MODE 0: store f32.  MODE 1: fused RoPE, store bf16 [h][s][128].
// MODE 2: store bf16 transposed [n][m] (i.e. [h*128+d][s]).
template <int MODE>
__global__ __launch_bounds__(128) void gemm_bf16(
    const unsigned short* __restrict__ A, const unsigned short* __restrict__ B,
    void* __restrict__ Dp, const float* __restrict__ cosT,
    const float* __restrict__ sinT) {
    __shared__ unsigned short bl[2][128 * 32];  // double-buffered B tile, 16 KB

    const int lane = threadIdx.x & 31;
    const int w    = threadIdx.x >> 5;
    const int ln   = lane & 15;     // A: row-in-tile / B: col / C: col
    const int hi   = lane >> 4;     // lane-half selects K chunk / row+8
    const int n0   = blockIdx.x * 128;
    const int m0   = blockIdx.y * 64;
    const int mrow = m0 + w * 16 + ln;

    const unsigned short* Bn = B + (size_t)n0 * DIM;
    const unsigned lds0 = (unsigned)(size_t)(&bl[0][0]);
    const unsigned lds1 = (unsigned)(size_t)(&bl[1][0]);

    v8f acc[8];
#pragma unroll
    for (int j = 0; j < 8; ++j)
#pragma unroll
        for (int e = 0; e < 8; ++e) acc[j][e] = 0.0f;

    if (w == 0) tdm_load_tile_128x32(Bn, lds0);  // prime buffer 0

    for (int k0 = 0; k0 < DIM; k0 += 32) {
        const int buf = (k0 >> 5) & 1;
        __syncthreads();  // all waves done computing from buf^1 (previous iter)
        if (w == 0) {
            if (k0 + 32 < DIM) {  // prefetch next tile into the other buffer
                tdm_load_tile_128x32(Bn + (k0 + 32), buf ? lds0 : lds1);
                __builtin_amdgcn_s_wait_tensorcnt(1);  // older load complete
            } else {
                __builtin_amdgcn_s_wait_tensorcnt(0);
            }
        }
        __syncthreads();  // tile `buf` visible to all waves

        Frag a;  // A fragment 16x32: lane=m, K chunks hi*8 and 16+hi*8
        {
            const unsigned short* ap = A + (size_t)mrow * DIM + k0 + hi * 8;
            a.u[0] = *(const uint4*)ap;
            a.u[1] = *(const uint4*)(ap + 16);
        }
#pragma unroll
        for (int nt = 0; nt < 8; ++nt) {
            Frag b;  // B^T rows (= weight rows) with the A lane pattern
            const unsigned short* bp = &bl[buf][(nt * 16 + ln) * 32 + hi * 8];
            b.u[0] = *(const uint4*)bp;
            b.u[1] = *(const uint4*)(bp + 16);
            acc[nt] = __builtin_amdgcn_wmma_f32_16x16x32_bf16(
                false, a.v, false, b.v, (short)0, acc[nt], false, false);
        }
    }

    // Epilogue. C layout: elem e -> row m0+w*16+e+8*hi ; col n0+nt*16+ln.
    if constexpr (MODE == 0) {
        float* D = (float*)Dp;
#pragma unroll
        for (int nt = 0; nt < 8; ++nt)
#pragma unroll
            for (int e = 0; e < 8; ++e) {
                int row = m0 + w * 16 + e + 8 * hi;
                int col = n0 + nt * 16 + ln;
                D[(size_t)row * DIM + col] = acc[nt][e];
            }
    } else if constexpr (MODE == 1) {
        // RoPE fused in registers: BN=128 head-aligned, pair dim is nt^4.
        unsigned short* D = (unsigned short*)Dp;
        const int h = n0 >> 7;
#pragma unroll
        for (int nt = 0; nt < 8; ++nt)
#pragma unroll
            for (int e = 0; e < 8; ++e) {
                int row  = m0 + w * 16 + e + 8 * hi;
                int dcol = nt * 16 + ln;
                float cs = cosT[row * HD + dcol];
                float sn = sinT[row * HD + dcol];
                float other = acc[nt ^ 4][e];          // value at d +/- 64
                float rot   = (dcol < 64) ? -other : other;
                float v     = acc[nt][e] * cs + rot * sn;
                D[((size_t)h * SEQ + row) * HD + dcol] = f2bf(v);
            }
    } else {  // MODE 2: V transposed -> [n][m]
        unsigned short* D = (unsigned short*)Dp;
#pragma unroll
        for (int nt = 0; nt < 8; ++nt)
#pragma unroll
            for (int e = 0; e < 8; ++e) {
                int row = m0 + w * 16 + e + 8 * hi;
                int col = n0 + nt * 16 + ln;
                D[(size_t)col * SEQ + row] = f2bf(acc[nt][e]);
            }
    }
}

// ---------------------------------------------------------------------------
// Flash attention: grid (SEQ/64, NH), 128 threads = 4 waves, 16 q-rows/wave.
__global__ __launch_bounds__(128) void attn_kernel(
    const unsigned short* __restrict__ Q,   // [h][s][128] bf16 (roped)
    const unsigned short* __restrict__ Kt,  // [h][s][128] bf16 (roped)
    const unsigned short* __restrict__ Vt,  // [h][128][s] bf16
    unsigned short* __restrict__ O) {       // [s][2048]   bf16
    __shared__ unsigned short pls[4][16 * 32];  // per-wave P transpose buffer

    const int lane = threadIdx.x & 31;
    const int w    = threadIdx.x >> 5;
    const int ln   = lane & 15;
    const int hi   = lane >> 4;
    const int h    = blockIdx.y;
    const int q0   = blockIdx.x * 64;
    const int qw   = q0 + w * 16;

    // Q fragments for this wave's 16 rows, hd = 4 chunks of 32 (A layout)
    Frag aq[4];
    {
        const unsigned short* qp = Q + ((size_t)h * SEQ + qw + ln) * HD + hi * 8;
#pragma unroll
        for (int c = 0; c < 4; ++c) {
            aq[c].u[0] = *(const uint4*)(qp + c * 32);
            aq[c].u[1] = *(const uint4*)(qp + c * 32 + 16);
        }
    }

    v8f   o[8];
    float mrow[8], lrow[8];
#pragma unroll
    for (int j = 0; j < 8; ++j)
#pragma unroll
        for (int e = 0; e < 8; ++e) o[j][e] = 0.0f;
#pragma unroll
    for (int e = 0; e < 8; ++e) { mrow[e] = -3.0e38f; lrow[e] = 0.0f; }

    const float kCap = 0.0017677669529663688f;  // (1/sqrt(128)) / 50
    const int   kend = q0 + 64;                 // causal upper bound

    for (int kb = 0; kb < kend; kb += 32) {
        if (kb + 32 < kend)
            __builtin_prefetch(Kt + ((size_t)h * SEQ + kb + 32 + ln) * HD, 0, 0);

        // ---- S = Q*K^T for 2 column tiles of 16 k-rows
        v8f st[2];
#pragma unroll
        for (int t = 0; t < 2; ++t) {
            v8f sa;
#pragma unroll
            for (int e = 0; e < 8; ++e) sa[e] = 0.0f;
            const unsigned short* kp =
                Kt + ((size_t)h * SEQ + kb + t * 16 + ln) * HD + hi * 8;
#pragma unroll
            for (int c = 0; c < 4; ++c) {
                Frag b;
                b.u[0] = *(const uint4*)(kp + c * 32);
                b.u[1] = *(const uint4*)(kp + c * 32 + 16);
                sa = __builtin_amdgcn_wmma_f32_16x16x32_bf16(
                    false, aq[c].v, false, b.v, (short)0, sa, false, false);
            }
            st[t] = sa;
        }

        // ---- scale + tanh soft-cap + causal mask
        float rm[8];
#pragma unroll
        for (int e = 0; e < 8; ++e) {
            int qr = qw + e + 8 * hi;
#pragma unroll
            for (int t = 0; t < 2; ++t) {
                float v = 50.0f * tanhf(st[t][e] * kCap);
                int  kc = kb + t * 16 + ln;
                if (kc > qr) v = -1.0e9f;
                st[t][e] = v;
            }
            rm[e] = fmaxf(st[0][e], st[1][e]);
        }
        // row-max across the 16 lanes of each half (rows live in vgpr index)
#pragma unroll
        for (int mk = 1; mk < 16; mk <<= 1)
#pragma unroll
            for (int e = 0; e < 8; ++e)
                rm[e] = fmaxf(rm[e], __shfl_xor(rm[e], mk, 32));

        // ---- online softmax rescale
#pragma unroll
        for (int e = 0; e < 8; ++e) {
            float mn = fmaxf(mrow[e], rm[e]);
            float sc = __expf(mrow[e] - mn);
            mrow[e]  = mn;
            lrow[e] *= sc;
#pragma unroll
            for (int j = 0; j < 8; ++j) o[j][e] *= sc;
        }

        // ---- P = exp(S - m), bf16, C-layout -> LDS -> A-layout (wave-private)
#pragma unroll
        for (int t = 0; t < 2; ++t)
#pragma unroll
            for (int e = 0; e < 8; ++e) {
                float p = __expf(st[t][e] - mrow[e]);
                lrow[e] += p;
                pls[w][(e + 8 * hi) * 32 + t * 16 + ln] = f2bf(p);
            }
        Frag ap;  // DS ops from one wave are in-order: RAW through LDS is safe
        {
            const unsigned short* pp = &pls[w][ln * 32 + hi * 8];
            ap.u[0] = *(const uint4*)pp;
            ap.u[1] = *(const uint4*)(pp + 16);
        }

        // ---- O += P * V  (V pre-transposed: contiguous along k)
#pragma unroll
        for (int j = 0; j < 8; ++j) {
            Frag bv;
            const unsigned short* vp =
                Vt + ((size_t)h * HD + j * 16 + ln) * SEQ + kb + hi * 8;
            bv.u[0] = *(const uint4*)vp;
            bv.u[1] = *(const uint4*)(vp + 16);
            o[j] = __builtin_amdgcn_wmma_f32_16x16x32_bf16(
                false, ap.v, false, bv.v, (short)0, o[j], false, false);
        }
    }

    // ---- finalize: reduce l across lanes, normalize, store bf16 [s][dim]
#pragma unroll
    for (int mk = 1; mk < 16; mk <<= 1)
#pragma unroll
        for (int e = 0; e < 8; ++e) lrow[e] += __shfl_xor(lrow[e], mk, 32);

#pragma unroll
    for (int e = 0; e < 8; ++e) {
        float inv = 1.0f / lrow[e];
        int   row = qw + e + 8 * hi;
#pragma unroll
        for (int j = 0; j < 8; ++j)
            O[(size_t)row * DIM + h * HD + j * 16 + ln] = f2bf(o[j][e] * inv);
    }
}

// ---------------------------------------------------------------------------
extern "C" void kernel_launch(void* const* d_in, const int* in_sizes, int n_in,
                              void* d_out, int out_size, void* d_ws,
                              size_t ws_size, hipStream_t stream) {
    (void)in_sizes; (void)n_in; (void)out_size; (void)ws_size;

    const float* x    = (const float*)d_in[0];
    const float* cosp = (const float*)d_in[1];
    const float* sinp = (const float*)d_in[2];
    // d_in[3] (mask) unused: causal mask applied analytically.
    const float* wq   = (const float*)d_in[4];
    const float* wk   = (const float*)d_in[5];
    const float* wv   = (const float*)d_in[6];
    const float* wo   = (const float*)d_in[7];

    char* ws = (char*)d_ws;
    const size_t MB8 = (size_t)SEQ * DIM * 2;  // 8 MB bf16 plane
    unsigned short* xb  = (unsigned short*)(ws + 0 * MB8);
    unsigned short* wqb = (unsigned short*)(ws + 1 * MB8);
    unsigned short* wkb = (unsigned short*)(ws + 2 * MB8);
    unsigned short* wvb = (unsigned short*)(ws + 3 * MB8);
    unsigned short* wob = (unsigned short*)(ws + 4 * MB8);
    unsigned short* qh  = (unsigned short*)(ws + 5 * MB8);  // [h][s][128]
    unsigned short* kh  = (unsigned short*)(ws + 6 * MB8);  // [h][s][128]
    unsigned short* vt  = (unsigned short*)(ws + 7 * MB8);  // [h][128][s]
    unsigned short* ob  = (unsigned short*)(ws + 8 * MB8);  // [s][2048]

    const int NELT = SEQ * DIM;
    const int cblk = (NELT + 255) / 256;
    cvt_f32_bf16<<<cblk, 256, 0, stream>>>(x,  xb,  NELT);
    cvt_f32_bf16<<<cblk, 256, 0, stream>>>(wq, wqb, NELT);
    cvt_f32_bf16<<<cblk, 256, 0, stream>>>(wk, wkb, NELT);
    cvt_f32_bf16<<<cblk, 256, 0, stream>>>(wv, wvb, NELT);
    cvt_f32_bf16<<<cblk, 256, 0, stream>>>(wo, wob, NELT);

    dim3 ggrid(DIM / 128, SEQ / 64);
    gemm_bf16<1><<<ggrid, 128, 0, stream>>>(xb, wqb, qh, cosp, sinp);
    gemm_bf16<1><<<ggrid, 128, 0, stream>>>(xb, wkb, kh, cosp, sinp);
    gemm_bf16<2><<<ggrid, 128, 0, stream>>>(xb, wvb, vt, nullptr, nullptr);

    attn_kernel<<<dim3(SEQ / 64, NH), 128, 0, stream>>>(qh, kh, vt, ob);

    gemm_bf16<0><<<ggrid, 128, 0, stream>>>(ob, wob, d_out, nullptr, nullptr);
}
